// ViT_ESM_30494267802093
// MI455X (gfx1250) — matrix operand order repaired and checked by reading, not compiled, not run on hardware
//
#include <hip/hip_runtime.h>
#include <hip/hip_bf16.h>
#include <hip/hip_fp16.h>

typedef __attribute__((ext_vector_type(16))) _Float16 v16h;
typedef __attribute__((ext_vector_type(8)))  float    v8f;

#define INFV 100000.0f

__device__ inline v8f wmma16(v16h a, v16h b, v8f c) {
  return __builtin_amdgcn_wmma_f32_16x16x32_f16(false, a, false, b, (short)0, c, false, false);
}
__device__ inline float geluf(float x)  { return 0.5f * x * (1.0f + erff(x * 0.70710678118f)); }
__device__ inline float leakyf(float x) { return x >= 0.f ? x : 0.2f * x; }

// ---------------------------------------------------------------- utilities
__global__ void __launch_bounds__(256) k_f2h(const float* s, _Float16* d, int n) {
  int i = blockIdx.x * 256 + threadIdx.x;
  if (i < n) d[i] = (_Float16)s[i];
}

// ------------------------------------------------- preprocess: mask + chem embed
__global__ void __launch_bounds__(256)
k_preproc(const float* feat, const float* xyz,
          const float* w1, const float* b1, const float* w2, const float* b2,
          float* xyzm, float* feats) {
  int i = blockIdx.x * 256 + threadIdx.x;            // point index over 2048*256
  if (i >= 2048 * 256) return;
  int b = i >> 8;
  float x0 = xyz[(size_t)b * 256 * 3 + 0], y0 = xyz[(size_t)b * 256 * 3 + 1], z0 = xyz[(size_t)b * 256 * 3 + 2];
  float px = xyz[(size_t)i * 3 + 0], py = xyz[(size_t)i * 3 + 1], pz = xyz[(size_t)i * 3 + 2];
  float dx = px - x0, dy = py - y0, dz = pz - z0;
  bool bad = sqrtf(dx * dx + dy * dy + dz * dz) > 20.0f;
  xyzm[(size_t)i * 3 + 0] = bad ? INFV : px;
  xyzm[(size_t)i * 3 + 1] = bad ? INFV : py;
  xyzm[(size_t)i * 3 + 2] = bad ? INFV : pz;
  float h[32];
  for (int o = 0; o < 32; o++) {
    float s = b1[o];
    for (int k = 0; k < 12; k++) s += feat[(size_t)i * 12 + k] * w1[k * 32 + o];
    h[o] = leakyf(s);
  }
  for (int o = 0; o < 32; o++) {
    float s = b2[o];
    for (int k = 0; k < 32; k++) s += h[k] * w2[k * 32 + o];
    feats[(size_t)i * 32 + o] = leakyf(s);
  }
}

// ------------------------------------------------- FPS + KNN, one block per cloud
__global__ void __launch_bounds__(256) k_fps(const float* xyzm, int* patches) {
  int b = blockIdx.x, t = threadIdx.x;
  __shared__ float px[256], py[256], pz[256], dmin[256], dcur[256];
  __shared__ float rv[256]; __shared__ int ri[256];
  px[t] = xyzm[((size_t)b * 256 + t) * 3 + 0];
  py[t] = xyzm[((size_t)b * 256 + t) * 3 + 1];
  pz[t] = xyzm[((size_t)b * 256 + t) * 3 + 2];
  bool bad = px[t] >= INFV * 0.5f;
  dmin[t] = bad ? -INFV : INFV;
  __syncthreads();
  int far = 0;
  for (int m = 0; m < 16; m++) {
    float cx = px[far], cy = py[far], cz = pz[far];
    float dx = px[t] - cx, dy = py[t] - cy, dz = pz[t] - cz;
    float d = dx * dx + dy * dy + dz * dz;
    dcur[t] = d; dmin[t] = fminf(dmin[t], d);
    rv[t] = dmin[t]; ri[t] = t;
    __syncthreads();
    for (int s = 128; s > 0; s >>= 1) {
      if (t < s) { if (rv[t + s] > rv[t]) { rv[t] = rv[t + s]; ri[t] = ri[t + s]; } }
      __syncthreads();
    }
    int newfar = ri[0];
    __syncthreads();
    for (int k = 0; k < 16; k++) {
      rv[t] = dcur[t]; ri[t] = t;
      __syncthreads();
      for (int s = 128; s > 0; s >>= 1) {
        if (t < s) {
          if (rv[t + s] < rv[t] || (rv[t + s] == rv[t] && ri[t + s] < ri[t])) { rv[t] = rv[t + s]; ri[t] = ri[t + s]; }
        }
        __syncthreads();
      }
      if (t == 0) { patches[((size_t)b * 16 + m) * 16 + k] = ri[0]; dcur[ri[0]] = 3.0e38f; }
      __syncthreads();
    }
    far = newfar;
  }
}

// ------------------------------------------------- patch centers & means
__global__ void __launch_bounds__(256) k_patchcenter(const float* xyzm, float* pcent) {
  int b = blockIdx.x, t = threadIdx.x;
  __shared__ float rx[256], ry[256], rz[256];
  rx[t] = xyzm[((size_t)b * 256 + t) * 3 + 0];
  ry[t] = xyzm[((size_t)b * 256 + t) * 3 + 1];
  rz[t] = xyzm[((size_t)b * 256 + t) * 3 + 2];
  __syncthreads();
  for (int s = 128; s > 0; s >>= 1) {
    if (t < s) { rx[t] += rx[t + s]; ry[t] += ry[t + s]; rz[t] += rz[t + s]; }
    __syncthreads();
  }
  if (t == 0) { pcent[b * 3 + 0] = rx[0] / 256.f; pcent[b * 3 + 1] = ry[0] / 256.f; pcent[b * 3 + 2] = rz[0] / 256.f; }
}

__global__ void __launch_bounds__(256) k_pcmean(const float* pcent, float* pcm) {
  int t = threadIdx.x;
  __shared__ float rx[256], ry[256], rz[256];
  float sx = 0, sy = 0, sz = 0;
  for (int i = t; i < 2048; i += 256) { sx += pcent[i * 3]; sy += pcent[i * 3 + 1]; sz += pcent[i * 3 + 2]; }
  rx[t] = sx; ry[t] = sy; rz[t] = sz;
  __syncthreads();
  for (int s = 128; s > 0; s >>= 1) {
    if (t < s) { rx[t] += rx[t + s]; ry[t] += ry[t + s]; rz[t] += rz[t + s]; }
    __syncthreads();
  }
  if (t == 0) { pcm[0] = rx[0] / 2048.f; pcm[1] = ry[0] / 2048.f; pcm[2] = rz[0] / 2048.f; }
}

__global__ void __launch_bounds__(256)
k_subc(const float* xyzm, const int* patches, const float* pcent, float* subc) {
  int i = blockIdx.x * 256 + threadIdx.x;            // (b, sp) row 0..32767
  if (i >= 32768) return;
  int b = i >> 4;
  float sx = 0, sy = 0, sz = 0;
  for (int j = 0; j < 16; j++) {
    int p = patches[(size_t)i * 16 + j];
    sx += xyzm[((size_t)b * 256 + p) * 3 + 0];
    sy += xyzm[((size_t)b * 256 + p) * 3 + 1];
    sz += xyzm[((size_t)b * 256 + p) * 3 + 2];
  }
  subc[(size_t)i * 3 + 0] = sx / 16.f - pcent[b * 3 + 0];
  subc[(size_t)i * 3 + 1] = sy / 16.f - pcent[b * 3 + 1];
  subc[(size_t)i * 3 + 2] = sz / 16.f - pcent[b * 3 + 2];
}

// ------------------------------------------------- gather f_in + LN(512) -> f16
__global__ void __launch_bounds__(256)
k_gather_ln(const float* feats, const int* patches, const float* g, const float* bv, _Float16* fin16) {
  int r = blockIdx.x;                                 // 0..32767
  int b = r >> 4;
  int t = threadIdx.x;
  __shared__ float buf[512]; __shared__ float red[256];
  for (int e = t; e < 512; e += 256) {
    int j = e >> 5, cc = e & 31;
    int p = patches[(size_t)r * 16 + j];
    buf[e] = feats[((size_t)b * 256 + p) * 32 + cc];
  }
  __syncthreads();
  float s = buf[t] + buf[t + 256];
  red[t] = s; __syncthreads();
  for (int st = 128; st > 0; st >>= 1) { if (t < st) red[t] += red[t + st]; __syncthreads(); }
  float mean = red[0] / 512.f; __syncthreads();
  float d0 = buf[t] - mean, d1 = buf[t + 256] - mean;
  red[t] = d0 * d0 + d1 * d1; __syncthreads();
  for (int st = 128; st > 0; st >>= 1) { if (t < st) red[t] += red[t + st]; __syncthreads(); }
  float inv = rsqrtf(red[0] / 512.f + 1e-5f); __syncthreads();
  for (int e = t; e < 512; e += 256)
    fin16[(size_t)r * 512 + e] = (_Float16)((buf[e] - mean) * inv * g[e] + bv[e]);
}

// ------------------------------------------------- generic row LayerNorm
__global__ void __launch_bounds__(256)
k_ln(const float* x, const float* g, const float* bv, float* outF, _Float16* outH, int W) {
  int r = blockIdx.x, t = threadIdx.x;
  __shared__ float red[256];
  float s = 0;
  for (int i = t; i < W; i += 256) s += x[(size_t)r * W + i];
  red[t] = s; __syncthreads();
  for (int st = 128; st > 0; st >>= 1) { if (t < st) red[t] += red[t + st]; __syncthreads(); }
  float mean = red[0] / W; __syncthreads();
  float v = 0;
  for (int i = t; i < W; i += 256) { float d = x[(size_t)r * W + i] - mean; v += d * d; }
  red[t] = v; __syncthreads();
  for (int st = 128; st > 0; st >>= 1) { if (t < st) red[t] += red[t + st]; __syncthreads(); }
  float inv = rsqrtf(red[0] / W + 1e-5f); __syncthreads();
  for (int i = t; i < W; i += 256) {
    float y = (x[(size_t)r * W + i] - mean) * inv * g[i] + bv[i];
    if (outF) outF[(size_t)r * W + i] = y;
    if (outH) outH[(size_t)r * W + i] = (_Float16)y;
  }
}

// ------------------------------------------------- templated WMMA GEMM
// out[M x N] = act(A[MxK] @ B[KxN] + bias) + resid ; ACT: 0 none, 1 gelu
template <int ACT, bool BIAS, bool RES, bool OF, bool OH>
__global__ void __launch_bounds__(256)
k_gemm_t(const _Float16* __restrict__ A, const _Float16* __restrict__ B,
         const float* __restrict__ bias, const float* __restrict__ resid,
         float* __restrict__ outF, _Float16* __restrict__ outH,
         int M, int N, int K) {
  int mt = blockIdx.x;
  int nbase = blockIdx.y * 128;
  int t = threadIdx.x, w = t >> 5, lane = t & 31;
  int ct = nbase + w * 16;
  __shared__ _Float16 sA[16 * 32];         // row-major 16 x 32
  __shared__ _Float16 sBt[128 * 40];       // col-major: [col][k], stride 40 (80B, 16B-aligned)
  v8f acc = {};
  for (int k0 = 0; k0 < K; k0 += 32) {
    __syncthreads();
    for (int i = t; i < 512; i += 256) {
      int r = i >> 5, c = i & 31;
      sA[i] = A[((size_t)mt * 16 + r) * K + k0 + c];
    }
    for (int i = t; i < 4096; i += 256) {
      int r = i >> 7, c = i & 127;
      int col = nbase + c;
      sBt[c * 40 + r] = (col < N) ? B[(size_t)(k0 + r) * N + col] : (_Float16)0.0f;
    }
    __syncthreads();
    if (ct < N) {
      v16h a, bf;
      int row = lane & 15, kb = (lane >> 4) << 4;
      const _Float16* ap = &sA[row * 32 + kb];
      #pragma unroll
      for (int e = 0; e < 8; e++) a[e] = ap[e];
      #pragma unroll
      for (int e = 0; e < 8; e++) a[8 + e] = ap[8 + e];
      int col = lane & 15, kb2 = (lane >> 4) << 4;
      const _Float16* bp = &sBt[(w * 16 + col) * 40 + kb2];
      #pragma unroll
      for (int e = 0; e < 16; e++) bf[e] = bp[e];
      acc = wmma16(a, bf, acc);
    }
  }
  if (ct >= N) return;
  int orow0 = mt * 16 + ((lane >> 4) << 3);
  int ocol = ct + (lane & 15);
  #pragma unroll
  for (int r = 0; r < 8; r++) {
    int gr = orow0 + r;
    float v = acc[r];
    if (BIAS) v += bias[ocol];
    if (ACT == 1) v = geluf(v);
    if (RES) v += resid[(size_t)gr * N + ocol];
    if (OF) outF[(size_t)gr * N + ocol] = v;
    if (OH) outH[(size_t)gr * N + ocol] = (_Float16)v;
  }
}

// ------------------------------------------------- small attention (seq 16, 8 heads, d=8)
__global__ void __launch_bounds__(128)
k_attn_small(const _Float16* qkv, _Float16* out) {
  int b = blockIdx.x, t = threadIdx.x;
  __shared__ _Float16 sq[16 * 64], sk[16 * 64], sv[16 * 64];
  for (int i = t; i < 16 * 192; i += 128) {
    int row = i / 192, c = i % 192;
    _Float16 v = qkv[((size_t)b * 16 + row) * 192 + c];
    if (c < 64) sq[row * 64 + c] = v;
    else if (c < 128) sk[row * 64 + (c - 64)] = v;
    else sv[row * 64 + (c - 128)] = v;
  }
  __syncthreads();
  int head = t >> 4, i = t & 15;
  float dots[16]; float mx = -3.0e38f;
  for (int j = 0; j < 16; j++) {
    float s = 0;
    for (int d = 0; d < 8; d++) s += (float)sq[i * 64 + head * 8 + d] * (float)sk[j * 64 + head * 8 + d];
    s *= 0.35355339f;
    dots[j] = s; mx = fmaxf(mx, s);
  }
  float sum = 0;
  for (int j = 0; j < 16; j++) { dots[j] = expf(dots[j] - mx); sum += dots[j]; }
  float o[8] = {0, 0, 0, 0, 0, 0, 0, 0};
  float inv = 1.f / sum;
  for (int j = 0; j < 16; j++) {
    float a = dots[j] * inv;
    for (int d = 0; d < 8; d++) o[d] += a * (float)sv[j * 64 + head * 8 + d];
  }
  for (int d = 0; d < 8; d++) out[((size_t)b * 16 + i) * 64 + head * 8 + d] = (_Float16)o[d];
}

// ------------------------------------------------- big attention (seq 2048), WMMA q.kT and attn.v
// dyn LDS: Sh (16x2048 f16, 64KB) | vlds (8x2048 f16, 32KB) | red (256 f32) | part (2048 f32)
__global__ void __launch_bounds__(256)
k_attn_big(const _Float16* __restrict__ qkv, _Float16* __restrict__ out) {
  extern __shared__ char smemraw[];
  _Float16* Sh   = (_Float16*)smemraw;
  _Float16* vlds = (_Float16*)(smemraw + 16 * 2048 * 2);
  float*    red  = (float*)(smemraw + 16 * 2048 * 2 + 8 * 2048 * 2);
  float*    part = red + 256;
  int head = blockIdx.y, qt = blockIdx.x;
  int t = threadIdx.x, w = t >> 5, lane = t & 31;
  int row = lane & 15;
  // stage this head's V d-major: vlds[d*2048 + j]
  for (int i = t; i < 2048 * 8; i += 256) {
    int j = i >> 3, d = i & 7;
    vlds[d * 2048 + j] = qkv[(size_t)j * 192 + 128 + head * 8 + d];
  }
  // q fragment (K padded 8 -> 32), scaled by 1/sqrt(8)
  v16h a;
  #pragma unroll
  for (int e = 0; e < 16; e++) a[e] = (_Float16)0.0f;
  if ((lane >> 4) == 0) {
    #pragma unroll
    for (int e = 0; e < 8; e++)
      a[e] = (_Float16)((float)qkv[((size_t)qt * 16 + row) * 192 + head * 8 + e] * 0.35355339f);
  }
  for (int jt = w; jt < 128; jt += 8) {
    v16h bf;
    int col = lane & 15;
    #pragma unroll
    for (int e = 0; e < 16; e++) bf[e] = (_Float16)0.0f;
    if ((lane >> 4) == 0) {
      #pragma unroll
      for (int e = 0; e < 8; e++)
        bf[e] = qkv[((size_t)jt * 16 + col) * 192 + 64 + head * 8 + e];
    }
    v8f c = {};
    c = wmma16(a, bf, c);
    #pragma unroll
    for (int r = 0; r < 8; r++)
      Sh[(((lane >> 4) << 3) + r) * 2048 + jt * 16 + col] = (_Float16)c[r];
  }
  __syncthreads();
  // softmax per row (f16 storage, f32 math)
  for (int r2 = 0; r2 < 16; r2++) {
    float m = -3.0e38f;
    for (int j = t; j < 2048; j += 256) m = fmaxf(m, (float)Sh[r2 * 2048 + j]);
    red[t] = m; __syncthreads();
    for (int s = 128; s > 0; s >>= 1) { if (t < s) red[t] = fmaxf(red[t], red[t + s]); __syncthreads(); }
    m = red[0]; __syncthreads();
    float sm = 0;
    for (int j = t; j < 2048; j += 256) {
      float e = expf((float)Sh[r2 * 2048 + j] - m);
      Sh[r2 * 2048 + j] = (_Float16)e; sm += e;
    }
    red[t] = sm; __syncthreads();
    for (int s = 128; s > 0; s >>= 1) { if (t < s) red[t] += red[t + s]; __syncthreads(); }
    float inv = 1.f / red[0]; __syncthreads();
    for (int j = t; j < 2048; j += 256)
      Sh[r2 * 2048 + j] = (_Float16)((float)Sh[r2 * 2048 + j] * inv);
    __syncthreads();
  }
  // out = S @ V   (N padded 8 -> 16); contiguous f16 fragment loads
  v8f acc = {};
  for (int kt = w; kt < 64; kt += 8) {
    v16h aa, bb;
    int kb = (lane >> 4) << 4;
    const _Float16* sp2 = &Sh[row * 2048 + kt * 32 + kb];
    #pragma unroll
    for (int e = 0; e < 8; e++) aa[e] = sp2[e];
    #pragma unroll
    for (int e = 0; e < 8; e++) aa[8 + e] = sp2[8 + e];
    int col = lane & 15, kb2 = (lane >> 4) << 4;
    if (col < 8) {
      const _Float16* vp = &vlds[col * 2048 + kt * 32 + kb2];
      #pragma unroll
      for (int e = 0; e < 16; e++) bb[e] = vp[e];
    } else {
      #pragma unroll
      for (int e = 0; e < 16; e++) bb[e] = (_Float16)0.0f;
    }
    acc = wmma16(aa, bb, acc);
  }
  #pragma unroll
  for (int r = 0; r < 8; r++)
    part[w * 256 + (((lane >> 4) << 3) + r) * 16 + (lane & 15)] = acc[r];
  __syncthreads();
  {
    int orow = t >> 4, ocol = t & 15;
    if (ocol < 8) {
      float s = 0;
      for (int ww = 0; ww < 8; ww++) s += part[ww * 256 + orow * 16 + ocol];
      out[((size_t)qt * 16 + orow) * 64 + head * 8 + ocol] = (_Float16)s;
    }
  }
}

// ------------------------------------------------- misc small kernels
__global__ void __launch_bounds__(256)
k_pos_h1(const float* cent, const float* sub, const float* w1, const float* b1, _Float16* h1, int R) {
  int i = blockIdx.x * 256 + threadIdx.x;
  if (i >= R * 128) return;
  int r = i >> 7, cc = i & 127;
  float c0 = cent[r * 3], c1 = cent[r * 3 + 1], c2 = cent[r * 3 + 2];
  if (sub) { c0 -= sub[0]; c1 -= sub[1]; c2 -= sub[2]; }
  float s = c0 * w1[cc] + c1 * w1[128 + cc] + c2 * w1[256 + cc] + b1[cc];
  h1[i] = (_Float16)geluf(s);
}

__global__ void __launch_bounds__(256) k_seqmax(const float* ln, float* subemb) {
  int i = blockIdx.x * 256 + threadIdx.x;            // 2048*128
  if (i >= 2048 * 128) return;
  int b = i >> 7, d = i & 127;
  float m = -3.0e38f;
  for (int sp = 0; sp < 16; sp++) m = fmaxf(m, ln[((size_t)b * 16 + sp) * 128 + d]);
  subemb[i] = m;
}

__global__ void __launch_bounds__(256)
k_buildx(const float* subemb, const float* posg, const float* esmemb, float* x0) {
  int i = blockIdx.x * 256 + threadIdx.x;            // 2048*256
  if (i >= 2048 * 256) return;
  int b = i >> 8, d = i & 255;
  x0[i] = (d < 128) ? (subemb[b * 128 + d] + posg[b * 128 + d]) : esmemb[b * 128 + (d - 128)];
}

__global__ void __launch_bounds__(128)
k_head(const float* xn, const float* subemb,
       const float* w1, const float* b1, const float* w2, const float* b2,
       const float* w3, const float* b3, float* out) {
  int b = blockIdx.x, t = threadIdx.x;
  __shared__ float xv[128], h1[16], h2[16];
  xv[t] = fmaxf(xn[(size_t)b * 256 + 2 * t], xn[(size_t)b * 256 + 2 * t + 1]) + subemb[(size_t)b * 128 + t];
  __syncthreads();
  if (t < 16) { float s = b1[t]; for (int i = 0; i < 128; i++) s += xv[i] * w1[i * 16 + t]; h1[t] = leakyf(s); }
  __syncthreads();
  if (t < 16) { float s = b2[t]; for (int i = 0; i < 16; i++) s += h1[i] * w2[i * 16 + t]; h2[t] = leakyf(s); }
  __syncthreads();
  if (t < 2)  { float s = b3[t]; for (int i = 0; i < 16; i++) s += h2[i] * w3[i * 2 + t]; out[(size_t)b * 2 + t] = s; }
}

// ================================================================ host side
extern "C" void kernel_launch(void* const* d_in, const int* in_sizes, int n_in,
                              void* d_out, int out_size, void* d_ws, size_t ws_size,
                              hipStream_t stream) {
  (void)in_sizes; (void)n_in; (void)out_size; (void)ws_size;
  const int N = 2048, R = 32768;                     // R = N * 16 sub-patch rows
  auto F = [&](int i) { return (const float*)d_in[i]; };
  // parameter indices (insertion order flattening of params dict)
  const int I_ce_w1 = 4, I_ce_b1 = 5, I_ce_w2 = 6, I_ce_b2 = 7;
  const int I_sp_ln1_g = 8, I_sp_ln1_b = 9, I_sp_w = 10, I_sp_b = 11, I_sp_ln2_g = 12, I_sp_ln2_b = 13;
  const int I_pos_w1 = 14, I_pos_b1 = 15, I_pos_w2 = 16, I_pos_b2 = 17;
  const int I_esm_w = 18, I_esm_b = 19, I_ln_g = 20, I_ln_b = 21;
  const int SL = 22;                                 // sub_layers: 4 x 11 tensors
  const int I_sub_norm_g = SL + 44, I_sub_norm_b = SL + 45;
  const int TL = SL + 46;                            // tf_layers: 4 x 11 tensors
  const int I_tf_norm_g = TL + 44, I_tf_norm_b = TL + 45;
  const int I_ph_w1 = TL + 46, I_ph_b1 = TL + 47, I_ph_w2 = TL + 48, I_ph_b2 = TL + 49,
            I_ph_w3 = TL + 50, I_ph_b3 = TL + 51;

  size_t off = 0;
  auto alloc = [&](size_t bytes) -> void* {
    void* p = (char*)d_ws + off;
    off += (bytes + 255) & ~(size_t)255;
    return p;
  };
  float*     xyzm    = (float*)alloc((size_t)N * 256 * 3 * 4);
  float*     feats   = (float*)alloc((size_t)N * 256 * 32 * 4);
  int*       patches = (int*)alloc((size_t)R * 16 * 4);
  float*     pcent   = (float*)alloc((size_t)N * 3 * 4);
  float*     pcm     = (float*)alloc(3 * 4);
  float*     subc    = (float*)alloc((size_t)R * 3 * 4);
  _Float16*  fin16   = (_Float16*)alloc((size_t)R * 512 * 2);
  float*     se      = (float*)alloc((size_t)R * 128 * 4);
  float*     lnbuf   = (float*)alloc((size_t)R * 128 * 4);
  _Float16*  h16     = (_Float16*)alloc((size_t)R * 256 * 2);
  _Float16*  g16     = (_Float16*)alloc((size_t)R * 512 * 2);
  _Float16*  qkv16   = (_Float16*)alloc((size_t)R * 192 * 2);
  _Float16*  ao16    = (_Float16*)alloc((size_t)R * 64 * 2);
  _Float16*  ph1     = (_Float16*)alloc((size_t)R * 128 * 2);
  float*     posg    = (float*)alloc((size_t)N * 128 * 4);
  _Float16*  esm16   = (_Float16*)alloc((size_t)N * 1280 * 2);
  float*     esmemb  = (float*)alloc((size_t)N * 128 * 4);
  float*     subemb  = (float*)alloc((size_t)N * 128 * 4);
  float*     x0      = (float*)alloc((size_t)N * 256 * 4);
  float*     X       = (float*)alloc((size_t)N * 256 * 4);
  float*     xn      = (float*)alloc((size_t)N * 256 * 4);
  _Float16*  spw16   = (_Float16*)alloc(512 * 128 * 2);
  _Float16*  pw216   = (_Float16*)alloc(128 * 128 * 2);
  _Float16*  esmw16  = (_Float16*)alloc(1280 * 128 * 2);
  _Float16 *swq[4], *swo[4], *sw1[4], *sw2[4], *twq[4], *two[4], *tw1[4], *tw2[4];
  for (int l = 0; l < 4; l++) {
    swq[l] = (_Float16*)alloc(128 * 192 * 2);  swo[l] = (_Float16*)alloc(64 * 128 * 2);
    sw1[l] = (_Float16*)alloc(128 * 512 * 2);  sw2[l] = (_Float16*)alloc(512 * 128 * 2);
  }
  for (int l = 0; l < 4; l++) {
    twq[l] = (_Float16*)alloc(256 * 192 * 2);  two[l] = (_Float16*)alloc(64 * 256 * 2);
    tw1[l] = (_Float16*)alloc(256 * 1024 * 2); tw2[l] = (_Float16*)alloc(1024 * 256 * 2);
  }

  auto cvt = [&](const float* src, _Float16* dst, int n) {
    k_f2h<<<(n + 255) / 256, 256, 0, stream>>>(src, dst, n);
  };
  cvt(F(I_sp_w), spw16, 512 * 128);
  cvt(F(I_pos_w2), pw216, 128 * 128);
  cvt(F(I_esm_w), esmw16, 1280 * 128);
  cvt((const float*)d_in[3], esm16, N * 1280);
  for (int l = 0; l < 4; l++) {
    int b0 = SL + 11 * l;
    cvt(F(b0 + 2), swq[l], 128 * 192); cvt(F(b0 + 3), swo[l], 64 * 128);
    cvt(F(b0 + 7), sw1[l], 128 * 512); cvt(F(b0 + 9), sw2[l], 512 * 128);
  }
  for (int l = 0; l < 4; l++) {
    int b0 = TL + 11 * l;
    cvt(F(b0 + 2), twq[l], 256 * 192); cvt(F(b0 + 3), two[l], 64 * 256);
    cvt(F(b0 + 7), tw1[l], 256 * 1024); cvt(F(b0 + 9), tw2[l], 1024 * 256);
  }

  // GEMM dispatch over the 4 epilogue variants actually used
  auto gemmF   = [&](const _Float16* A, const _Float16* B, const float* bias,
                     float* outF, int M, int Nn, int K) {       // bias, f32 out
    k_gemm_t<0, true, false, true, false>
      <<<dim3(M / 16, (Nn + 127) / 128), 256, 0, stream>>>(A, B, bias, nullptr, outF, nullptr, M, Nn, K);
  };
  auto gemmFR  = [&](const _Float16* A, const _Float16* B, const float* bias,
                     const float* resid, float* outF, int M, int Nn, int K) { // bias+resid, f32 out
    k_gemm_t<0, true, true, true, false>
      <<<dim3(M / 16, (Nn + 127) / 128), 256, 0, stream>>>(A, B, bias, resid, outF, nullptr, M, Nn, K);
  };
  auto gemmH   = [&](const _Float16* A, const _Float16* B, _Float16* outH, int M, int Nn, int K) { // plain, f16 out
    k_gemm_t<0, false, false, false, true>
      <<<dim3(M / 16, (Nn + 127) / 128), 256, 0, stream>>>(A, B, nullptr, nullptr, nullptr, outH, M, Nn, K);
  };
  auto gemmHG  = [&](const _Float16* A, const _Float16* B, const float* bias,
                     _Float16* outH, int M, int Nn, int K) {    // bias+gelu, f16 out
    k_gemm_t<1, true, false, false, true>
      <<<dim3(M / 16, (Nn + 127) / 128), 256, 0, stream>>>(A, B, bias, nullptr, nullptr, outH, M, Nn, K);
  };

  // ---- stage 1: preprocess, FPS/KNN, centers, gather+LN
  k_preproc<<<(N * 256) / 256, 256, 0, stream>>>(F(0), F(1), F(I_ce_w1), F(I_ce_b1), F(I_ce_w2), F(I_ce_b2), xyzm, feats);
  k_fps<<<N, 256, 0, stream>>>(xyzm, patches);
  k_patchcenter<<<N, 256, 0, stream>>>(xyzm, pcent);
  k_pcmean<<<1, 256, 0, stream>>>(pcent, pcm);
  k_subc<<<R / 256, 256, 0, stream>>>(xyzm, patches, pcent, subc);
  k_gather_ln<<<R, 256, 0, stream>>>(feats, patches, F(I_sp_ln1_g), F(I_sp_ln1_b), fin16);

  // ---- stage 2: sub-patch embedding: se = LN2(fin@sp_w+sp_b) + pos(subc)
  gemmF(fin16, spw16, F(I_sp_b), se, R, 128, 512);
  k_ln<<<R, 256, 0, stream>>>(se, F(I_sp_ln2_g), F(I_sp_ln2_b), lnbuf, nullptr, 128);
  k_pos_h1<<<(R * 128) / 256, 256, 0, stream>>>(subc, nullptr, F(I_pos_w1), F(I_pos_b1), ph1, R);
  gemmFR(ph1, pw216, F(I_pos_b2), lnbuf, se, R, 128, 128);

  // ---- transformer layer runner
  auto run_layer = [&](float* x, int Mr, int D, const _Float16* wq, const _Float16* wo,
                       const _Float16* w1, const _Float16* w2, int pb, int FFN, bool big) {
    k_ln<<<Mr, 256, 0, stream>>>(x, F(pb + 0), F(pb + 1), nullptr, h16, D);
    gemmH(h16, wq, qkv16, Mr, 192, D);
    if (big) {
      size_t sm = (size_t)16 * 2048 * 2 + (size_t)8 * 2048 * 2 + (size_t)(256 + 2048) * 4;
      k_attn_big<<<dim3(128, 8), 256, sm, stream>>>(qkv16, ao16);
    } else {
      k_attn_small<<<Mr / 16, 128, 0, stream>>>(qkv16, ao16);
    }
    gemmFR(ao16, wo, F(pb + 4), x, x, Mr, D, 64);
    k_ln<<<Mr, 256, 0, stream>>>(x, F(pb + 5), F(pb + 6), nullptr, h16, D);
    gemmHG(h16, w1, F(pb + 8), g16, Mr, FFN, D);
    gemmFR(g16, w2, F(pb + 10), x, x, Mr, D, FFN);
  };

  // ---- stage 3: sub transformer (seq 16, dim 128) + norm + seq max
  for (int l = 0; l < 4; l++)
    run_layer(se, R, 128, swq[l], swo[l], sw1[l], sw2[l], SL + 11 * l, 512, false);
  k_ln<<<R, 256, 0, stream>>>(se, F(I_sub_norm_g), F(I_sub_norm_b), lnbuf, nullptr, 128);
  k_seqmax<<<(N * 128) / 256, 256, 0, stream>>>(lnbuf, subemb);

  // ---- stage 4: global input: esm proj, pos(pc), concat, LN
  gemmF(esm16, esmw16, F(I_esm_b), esmemb, N, 128, 1280);
  k_pos_h1<<<(N * 128) / 256, 256, 0, stream>>>(pcent, pcm, F(I_pos_w1), F(I_pos_b1), ph1, N);
  gemmF(ph1, pw216, F(I_pos_b2), posg, N, 128, 128);
  k_buildx<<<(N * 256) / 256, 256, 0, stream>>>(subemb, posg, esmemb, x0);
  k_ln<<<N, 256, 0, stream>>>(x0, F(I_ln_g), F(I_ln_b), X, nullptr, 256);

  // ---- stage 5: global transformer (seq 2048, dim 256) + norm
  for (int l = 0; l < 4; l++)
    run_layer(X, N, 256, twq[l], two[l], tw1[l], tw2[l], TL + 11 * l, 1024, true);
  k_ln<<<N, 256, 0, stream>>>(X, F(I_tf_norm_g), F(I_tf_norm_b), xn, nullptr, 256);

  // ---- stage 6: pair-max pool + head
  k_head<<<N, 128, 0, stream>>>(xn, subemb, F(I_ph_w1), F(I_ph_b1), F(I_ph_w2), F(I_ph_b2),
                                F(I_ph_w3), F(I_ph_b3), (float*)d_out);
}
